// MultiHeadAttention_48206712930585
// MI455X (gfx1250) — matrix operand order repaired and checked by reading, not compiled, not run on hardware
//
#include <hip/hip_runtime.h>

#define BB 2
#define SS 2048
#define DD 1024
#define NH 16
#define HDIM 64

typedef __attribute__((ext_vector_type(16))) _Float16 v16h;
typedef __attribute__((ext_vector_type(8)))  _Float16 v8h;
typedef __attribute__((ext_vector_type(8)))  float    v8f;

// ---------- WMMA operand helpers (wave32, v_wmma_f32_16x16x32_f16) ----------
// A 16x32 f16: lane holds row m=lane&15; lanes0-15 elems = K0..7,K16..23,
//              lanes16-31 elems = K8..15,K24..31.
// B 32x16 f16: lane holds col n=lane&15; same K order as A.
// C/D 16x16 f32: lane = col n=lane&15; vgpr j = row (lane>>4)*8+j.
__device__ __forceinline__ v16h cat16(v8h lo, v8h hi) {
  v16h r;
#pragma unroll
  for (int i = 0; i < 8; ++i) { r[i] = lo[i]; r[i + 8] = hi[i]; }
  return r;
}
__device__ __forceinline__ v16h load_a32(const _Float16* row, int lh) {
  v8h lo = *(const v8h*)(row + lh * 8);
  v8h hi = *(const v8h*)(row + 16 + lh * 8);
  return cat16(lo, hi);
}
__device__ __forceinline__ v16h load_b32(const _Float16* colk, int lh) {
  v8h lo = *(const v8h*)(colk + lh * 8);
  v8h hi = *(const v8h*)(colk + 16 + lh * 8);
  return cat16(lo, hi);
}
__device__ __forceinline__ v8f wmma16(v16h a, v16h b, v8f c) {
  return __builtin_amdgcn_wmma_f32_16x16x32_f16(false, a, false, b, (short)0, c,
                                                false, false);
}
__device__ __forceinline__ float redmax16(float v) {
#pragma unroll
  for (int m = 1; m < 16; m <<= 1) v = fmaxf(v, __shfl_xor(v, m, 32));
  return v;
}
__device__ __forceinline__ float redsum16(float v) {
#pragma unroll
  for (int m = 1; m < 16; m <<= 1) v += __shfl_xor(v, m, 32);
  return v;
}

// Async global->LDS 16B copy (CDNA5, tracked by ASYNCcnt). lds = addrspace(3)
// byte offset (low 32 bits of generic pointer), voff = byte offset from base.
__device__ __forceinline__ void async_cp16(uint32_t lds, uint64_t base,
                                           uint32_t voff) {
  asm volatile("global_load_async_to_lds_b128 %0, %1, %2"
               :: "v"(lds), "v"(voff), "s"(base) : "memory");
}
__device__ __forceinline__ void wait_async0() {
  asm volatile("s_wait_asynccnt 0x0" ::: "memory");
}
__device__ __forceinline__ uint32_t lds_off(const void* p) {
  return (uint32_t)(uintptr_t)p;  // generic LDS address: low 32 bits = offset
}

// ---------------- prep: x fp32 -> f16 --------------------------------------
__global__ __launch_bounds__(256) void cvt_kernel(const float* __restrict__ in,
                                                  _Float16* __restrict__ out) {
  const size_t i = ((size_t)blockIdx.x * 256 + threadIdx.x) * 8;
  const float4 a = *(const float4*)(in + i);
  const float4 b = *(const float4*)(in + i + 4);
  v8h o;
  o[0] = (_Float16)a.x; o[1] = (_Float16)a.y;
  o[2] = (_Float16)a.z; o[3] = (_Float16)a.w;
  o[4] = (_Float16)b.x; o[5] = (_Float16)b.y;
  o[6] = (_Float16)b.z; o[7] = (_Float16)b.w;
  *(v8h*)(out + i) = o;
}

// ---------------- prep: W [1024][N] fp32 -> Wt [N][1024] f16 (LDS tiled) ----
__global__ __launch_bounds__(256) void transpose_kernel(
    const float* __restrict__ W, _Float16* __restrict__ Wt, int N) {
  __shared__ _Float16 t[64][72];  // padded to dodge bank conflicts
  const int tid = threadIdx.x;
  const int k0 = blockIdx.x * 64;
  const int n0 = blockIdx.y * 64;
#pragma unroll
  for (int i = 0; i < 4; ++i) {  // read 64x64 fp32, coalesced along n
    int c = tid + i * 256;
    int row = c >> 4;
    int col = (c & 15) << 2;
    const float4 v = *(const float4*)(W + (size_t)(k0 + row) * N + n0 + col);
    t[row][col + 0] = (_Float16)v.x;
    t[row][col + 1] = (_Float16)v.y;
    t[row][col + 2] = (_Float16)v.z;
    t[row][col + 3] = (_Float16)v.w;
  }
  __syncthreads();
#pragma unroll
  for (int i = 0; i < 2; ++i) {  // write transposed, coalesced along k
    int c = tid + i * 256;
    int nrow = c >> 3;
    int kcol = (c & 7) << 3;
    v8h o;
#pragma unroll
    for (int e = 0; e < 8; ++e) o[e] = t[kcol + e][nrow];
    *(v8h*)(Wt + (size_t)(n0 + nrow) * DD + k0 + kcol) = o;
  }
}

// ---------------- shared GEMM core: C[64x128] += A[64xK] * Bt[128xK]^T ------
// A: f16 [4096][1024] row-major; Bt: f16 [N][1024] row-major (pre-transposed).
// 8 waves, wave tile 32x32, K-stage 64, LDS double-buffered async pipeline.
__device__ __forceinline__ void gemm_core(const _Float16* __restrict__ A,
                                          const _Float16* __restrict__ Bt,
                                          int m0, int n0, int tid,
                                          v8f acc[2][2]) {
  __shared__ _Float16 xs[2][64][64];
  __shared__ _Float16 wt[2][128][64];
  const int lane = tid & 31;
  const int ln = lane & 15, lh = lane >> 4;
  const int wave = tid >> 5;
  const int wm = (wave >> 2) * 32;
  const int wn = (wave & 3) * 32;
  const uint64_t abase = (uint64_t)(uintptr_t)A;
  const uint64_t bbase = (uint64_t)(uintptr_t)Bt;

  auto issue = [&](int k0, int buf) {
#pragma unroll
    for (int i = 0; i < 2; ++i) {  // A tile 64x64 f16 = 512 x 16B chunks
      int c = tid + i * 256;
      int row = c >> 3, col = (c & 7) << 3;
      async_cp16(lds_off(&xs[buf][row][col]), abase,
                 (uint32_t)(((m0 + row) * DD + k0 + col) * 2));
    }
#pragma unroll
    for (int i = 0; i < 4; ++i) {  // Bt tile 128x64 f16 = 1024 x 16B chunks
      int c = tid + i * 256;
      int row = c >> 3, col = (c & 7) << 3;
      async_cp16(lds_off(&wt[buf][row][col]), bbase,
                 (uint32_t)(((n0 + row) * DD + k0 + col) * 2));
    }
  };

  issue(0, 0);
  int buf = 0;
  for (int ks = 0; ks < DD / 64; ++ks) {
    wait_async0();       // this wave's stage-ks copies have landed in LDS
    __syncthreads();     // everyone's copies visible; prev compute finished
    if (ks + 1 < DD / 64) issue((ks + 1) * 64, buf ^ 1);  // overlap next stage
#pragma unroll
    for (int kk = 0; kk < 64; kk += 32) {
      v16h a0 = load_a32(&xs[buf][wm + ln][kk], lh);
      v16h a1 = load_a32(&xs[buf][wm + 16 + ln][kk], lh);
      v16h b0 = load_b32(&wt[buf][wn + ln][kk], lh);
      v16h b1 = load_b32(&wt[buf][wn + 16 + ln][kk], lh);
      acc[0][0] = wmma16(a0, b0, acc[0][0]);
      acc[0][1] = wmma16(a0, b1, acc[0][1]);
      acc[1][0] = wmma16(a1, b0, acc[1][0]);
      acc[1][1] = wmma16(a1, b1, acc[1][1]);
    }
    buf ^= 1;
  }
}

// ---------------- Kernel A: qkv GEMM + scatter to Q/K/Vt f16 ---------------
__global__ __launch_bounds__(256) void qkv_kernel(
    const _Float16* __restrict__ xh, const _Float16* __restrict__ Wt,
    const float* __restrict__ bqkv, _Float16* __restrict__ qh,
    _Float16* __restrict__ kh, _Float16* __restrict__ vth) {
  const int tid = threadIdx.x;
  const int lane = tid & 31;
  const int ln = lane & 15, lh = lane >> 4;
  const int wave = tid >> 5;
  const int wm = (wave >> 2) * 32;
  const int wn = (wave & 3) * 32;
  const int m0 = blockIdx.y * 64;
  const int n0 = blockIdx.x * 128;

  v8f acc[2][2] = {};
  gemm_core(xh, Wt, m0, n0, tid, acc);

  const int bcol = m0 >> 11;               // batch constant per block
  const int sbase = (m0 & (SS - 1)) + wm + lh * 8;
#pragma unroll
  for (int sn = 0; sn < 2; ++sn) {
    const int ng = n0 + wn + sn * 16 + ln;  // column in [0, 3072)
    const float bias = bqkv[ng];
    const int h = ng / (3 * HDIM);
    const int r = ng % (3 * HDIM);
    const size_t bh = (size_t)(bcol * NH + h);
    _Float16* dst;
    int sstr;
    if (r < HDIM) {
      dst = qh + bh * SS * HDIM + r;                      sstr = HDIM;
    } else if (r < 2 * HDIM) {
      dst = kh + bh * SS * HDIM + (r - HDIM);             sstr = HDIM;
    } else {
      dst = vth + (bh * HDIM + (r - 2 * HDIM)) * SS;      sstr = 1;
    }
#pragma unroll
    for (int sm = 0; sm < 2; ++sm)
#pragma unroll
      for (int j = 0; j < 8; ++j)
        dst[(size_t)(sbase + sm * 16 + j) * sstr] =
            (_Float16)(acc[sm][sn][j] + bias);
  }
}

// ---------------- Kernel B: causal flash attention --------------------------
// Grid (S/64, B*H), block 128 (4 waves). Wave: 16 queries x full HD=64.
__global__ __launch_bounds__(128) void attn_kernel(
    const _Float16* __restrict__ qh, const _Float16* __restrict__ kh,
    const _Float16* __restrict__ vth, _Float16* __restrict__ ao) {
  __shared__ _Float16 pl[4][16][32];  // per-wave P transpose scratch
  const int tid = threadIdx.x;
  const int wave = tid >> 5, lane = tid & 31;
  const int ln = lane & 15, lh = lane >> 4;
  const int bh = blockIdx.y;
  const int b = bh >> 4, h = bh & (NH - 1);
  const int q0 = blockIdx.x * 64 + wave * 16;

  const _Float16* qrow = qh + ((size_t)bh * SS + q0 + ln) * HDIM;
  const v16h qa0 = load_a32(qrow, lh);
  const v16h qa1 = load_a32(qrow + 32, lh);

  v8f o[4] = {};
  float mrow[8], lrow[8];
#pragma unroll
  for (int j = 0; j < 8; ++j) { mrow[j] = -1e30f; lrow[j] = 0.0f; }
  const float scale = 0.125f;  // 1/sqrt(HD)

  for (int k0 = 0; k0 < q0 + 16; k0 += 32) {  // causal extent, 32-key blocks
    if (k0 + 32 < q0 + 16) {  // prefetch next K/V block (global_prefetch_b8)
      __builtin_prefetch(kh + ((size_t)bh * SS + k0 + 32 + ln) * HDIM, 0, 3);
      __builtin_prefetch(vth + ((size_t)bh * HDIM + ln) * SS + k0 + 32, 0, 3);
    }
    v8f sc[2];
#pragma unroll
    for (int t = 0; t < 2; ++t) {
      const _Float16* krow = kh + ((size_t)bh * SS + k0 + t * 16 + ln) * HDIM;
      v8f c = {};
      c = wmma16(qa0, load_b32(krow, lh), c);
      c = wmma16(qa1, load_b32(krow + 32, lh), c);
      sc[t] = c;
    }
    float p0v[8], p1v[8], alpha[8];
#pragma unroll
    for (int j = 0; j < 8; ++j) {
      const int qg = q0 + lh * 8 + j;
      float s0 = sc[0][j] * scale + ((k0 + ln) <= qg ? 0.0f : -1e30f);
      float s1 = sc[1][j] * scale + ((k0 + 16 + ln) <= qg ? 0.0f : -1e30f);
      const float rm = redmax16(fmaxf(s0, s1));
      const float mn = fmaxf(mrow[j], rm);
      alpha[j] = __expf(mrow[j] - mn);
      mrow[j] = mn;
      const float p0 = __expf(s0 - mn);
      const float p1 = __expf(s1 - mn);
      p0v[j] = p0;
      p1v[j] = p1;
      lrow[j] = lrow[j] * alpha[j] + redsum16(p0 + p1);
    }
#pragma unroll
    for (int dt = 0; dt < 4; ++dt)
#pragma unroll
      for (int j = 0; j < 8; ++j) o[dt][j] *= alpha[j];

    // D-layout P -> LDS -> A-operand (wave-internal; LDS in-order per wave)
#pragma unroll
    for (int j = 0; j < 8; ++j) {
      pl[wave][lh * 8 + j][ln] = (_Float16)p0v[j];
      pl[wave][lh * 8 + j][16 + ln] = (_Float16)p1v[j];
    }
    asm volatile("" ::: "memory");
    const v16h pa = load_a32(&pl[wave][ln][0], lh);
#pragma unroll
    for (int dt = 0; dt < 4; ++dt) {
      const _Float16* vrow = vth + ((size_t)bh * HDIM + dt * 16 + ln) * SS + k0;
      o[dt] = wmma16(pa, load_b32(vrow, lh), o[dt]);
    }
  }

#pragma unroll
  for (int j = 0; j < 8; ++j) lrow[j] = 1.0f / lrow[j];
#pragma unroll
  for (int dt = 0; dt < 4; ++dt)
#pragma unroll
    for (int j = 0; j < 8; ++j) {
      const int qg = q0 + lh * 8 + j;
      const int d = h * HDIM + dt * 16 + ln;
      ao[((size_t)b * SS + qg) * DD + d] = (_Float16)(o[dt][j] * lrow[j]);
    }
}

// ---------------- Kernel C: out = vals @ W_o + b_o --------------------------
__global__ __launch_bounds__(256) void proj_kernel(
    const _Float16* __restrict__ ao, const _Float16* __restrict__ Wt,
    const float* __restrict__ bo, float* __restrict__ out) {
  const int tid = threadIdx.x;
  const int lane = tid & 31;
  const int ln = lane & 15, lh = lane >> 4;
  const int wave = tid >> 5;
  const int wm = (wave >> 2) * 32;
  const int wn = (wave & 3) * 32;
  const int m0 = blockIdx.y * 64;
  const int n0 = blockIdx.x * 128;

  v8f acc[2][2] = {};
  gemm_core(ao, Wt, m0, n0, tid, acc);

#pragma unroll
  for (int sn = 0; sn < 2; ++sn) {
    const int ng = n0 + wn + sn * 16 + ln;
    const float bias = bo[ng];
#pragma unroll
    for (int sm = 0; sm < 2; ++sm)
#pragma unroll
      for (int j = 0; j < 8; ++j) {
        const int mg = m0 + wm + sm * 16 + lh * 8 + j;
        out[(size_t)mg * DD + ng] = acc[sm][sn][j] + bias;
      }
  }
}

extern "C" void kernel_launch(void* const* d_in, const int* in_sizes, int n_in,
                              void* d_out, int out_size, void* d_ws,
                              size_t ws_size, hipStream_t stream) {
  (void)in_sizes; (void)n_in; (void)out_size; (void)ws_size;
  const float* x = (const float*)d_in[0];
  // d_in[1] = additive mask: unused, causal mask applied analytically
  const float* Wqkv = (const float*)d_in[2];
  const float* bqkv = (const float*)d_in[3];
  const float* Wo = (const float*)d_in[4];
  const float* bo = (const float*)d_in[5];
  float* out = (float*)d_out;

  const size_t M = (size_t)BB * SS;        // 4096
  _Float16* xh    = (_Float16*)d_ws;       // [4096][1024]      8 MB
  _Float16* wqkvt = xh + M * DD;           // [3072][1024]      6 MB
  _Float16* wot   = wqkvt + 3 * DD * DD;   // [1024][1024]      2 MB
  _Float16* qh    = wot + DD * DD;         // [B*H][S][HD]      8 MB
  _Float16* kh    = qh + M * DD;           //                   8 MB
  _Float16* vth   = kh + M * DD;           // [B*H][HD][S]      8 MB
  _Float16* ao    = vth + M * DD;          // [B][S][D]         8 MB

  cvt_kernel<<<dim3(M * DD / 2048), 256, 0, stream>>>(x, xh);
  transpose_kernel<<<dim3(DD / 64, 3 * DD / 64), 256, 0, stream>>>(Wqkv, wqkvt,
                                                                   3 * DD);
  transpose_kernel<<<dim3(DD / 64, DD / 64), 256, 0, stream>>>(Wo, wot, DD);
  qkv_kernel<<<dim3(3 * DD / 128, M / 64), 256, 0, stream>>>(xh, wqkvt, bqkv,
                                                             qh, kh, vth);
  attn_kernel<<<dim3(SS / 64, BB * NH), 128, 0, stream>>>(qh, kh, vth, ao);
  proj_kernel<<<dim3(DD / 128, M / 64), 256, 0, stream>>>(ao, wot, bo, out);
}
